// LoRAQKVParallelLinear_22101901705622
// MI455X (gfx1250) — matrix operand order repaired and checked by reading, not compiled, not run on hardware
//
#include <hip/hip_runtime.h>
#include <hip/hip_bf16.h>
#include <stddef.h>
#include <stdint.h>

// ---- plain ext-vector types (trivially constructible, union-safe) ----
typedef __attribute__((ext_vector_type(16))) __bf16       v16bf;
typedef __attribute__((ext_vector_type(4)))  __bf16       v4bf;
typedef __attribute__((ext_vector_type(8)))  float        v8f;
typedef __attribute__((ext_vector_type(4)))  float        f32x4;
typedef __attribute__((ext_vector_type(4)))  unsigned int u32x4;

#define T_TOK 8192
#define HID   2048
#define QSZ   2048
#define KVSZ  512
#define OUTN  3072
#define LRANK 16

#define BM 128
#define BN 128
#define BK 32
#define LDT (BK + 8)          // padded LDS stride (bf16 elems): 40 -> 20-bank row rotation
#define NBUF 3                // 3-deep async pipeline
#define BUFSTRIDE (BM * LDT * 2)   // bytes per buffer per matrix (10240)

// ================= fp32 -> bf16 conversion =================
__global__ void cvt_bf16_kernel(const float* __restrict__ src,
                                __bf16* __restrict__ dst, int n4) {
  int i = blockIdx.x * blockDim.x + threadIdx.x;
  if (i >= n4) return;
  f32x4 f = ((const f32x4*)src)[i];
  v4bf o;
  o.x = (__bf16)f.x; o.y = (__bf16)f.y; o.z = (__bf16)f.z; o.w = (__bf16)f.w;
  ((v4bf*)dst)[i] = o;
}

// ================= per-token LoRA activations ==============
// h_scaled[t, g, r] = scaling[slot_t] * dot(lora_A[slot_t, g, r, :], x[t, :])
__global__ void lora_h_kernel(const float* __restrict__ x,
                              const float* __restrict__ loraA,
                              const float* __restrict__ scaling,
                              const int*   __restrict__ slots,
                              float* __restrict__ hbuf) {
  const int t    = blockIdx.x;
  const int lane = threadIdx.x & 31;
  const int w    = threadIdx.x >> 5;          // 8 waves, 6 outputs each (48 total)
  const int slot = slots[t];
  const float sc = scaling[slot];
  const f32x4* xr = (const f32x4*)(x + (size_t)t * HID);
  for (int o = w * 6; o < w * 6 + 6; ++o) {
    const int g = o >> 4, r = o & 15;
    const f32x4* ar = (const f32x4*)(loraA + ((((size_t)slot * 3 + g) * LRANK) + r) * HID);
    float s = 0.f;
    for (int c = lane; c < HID / 4; c += 32) {
      f32x4 xv = xr[c], av = ar[c];
      s += xv.x * av.x + xv.y * av.y + xv.z * av.z + xv.w * av.w;
    }
    for (int off = 16; off; off >>= 1) s += __shfl_xor(s, off, 32);
    if (lane == 0) hbuf[(size_t)t * 48 + o] = s * sc;
  }
}

// ================= WMMA GEMM + fused LoRA epilogue =========
union AFrag { u32x4 q[2]; v16bf v; };

__global__ __launch_bounds__(256)
void qkv_gemm_kernel(const __bf16* __restrict__ xb,
                     const __bf16* __restrict__ wb,
                     const float*  __restrict__ hbuf,
                     const float*  __restrict__ Bq,
                     const float*  __restrict__ Bk,
                     const float*  __restrict__ Bv,
                     const int*    __restrict__ slots,
                     float* __restrict__ out) {
  __shared__ __bf16 sA[NBUF][BM * LDT];
  __shared__ __bf16 sB[NBUF][BN * LDT];

  const int tid  = threadIdx.x;
  const int lane = tid & 31;
  const int w    = tid >> 5;
  const int wm   = w >> 2;          // 0..1  (M waves)
  const int wn   = w & 3;           // 0..3  (N waves)
  const int m0   = blockIdx.y * BM;
  const int n0   = blockIdx.x * BN;

  v8f acc[4][2];
  const v8f vzero = {0.f, 0.f, 0.f, 0.f, 0.f, 0.f, 0.f, 0.f};
#pragma unroll
  for (int i = 0; i < 4; ++i)
#pragma unroll
    for (int j = 0; j < 2; ++j) acc[i][j] = vzero;

  // ---- async staging: 256 threads, each owns a 32B slice of A and of B per K-tile
  const int srow = tid >> 1;            // 0..127
  const int scol = (tid & 1) * 16;      // 0 or 16 (bf16 elems)
  const uint64_t gA0 = (uint64_t)(uintptr_t)(xb + (size_t)(m0 + srow) * HID + scol);
  const uint64_t gB0 = (uint64_t)(uintptr_t)(wb + (size_t)(n0 + srow) * HID + scol);
  // CDNA5 flat->LDS aperture: addr[63:32] selects LDS space, addr[31:0] is the LDS address
  const uint32_t ldsA0 = (uint32_t)(uintptr_t)(&sA[0][0]) + (uint32_t)(srow * LDT + scol) * 2u;
  const uint32_t ldsB0 = (uint32_t)(uintptr_t)(&sB[0][0]) + (uint32_t)(srow * LDT + scol) * 2u;

  // 4 async b128 copies per thread per K-tile; offset: applies to BOTH global and LDS addr
  auto issue_tile = [&](int kt, int buf) {
    const uint64_t ga = gA0 + (uint64_t)kt * (BK * 2);
    const uint64_t gb = gB0 + (uint64_t)kt * (BK * 2);
    const uint32_t la = ldsA0 + (uint32_t)buf * BUFSTRIDE;
    const uint32_t lb = ldsB0 + (uint32_t)buf * BUFSTRIDE;
    asm volatile("global_load_async_to_lds_b128 %0, %1, off\n\t"
                 "global_load_async_to_lds_b128 %0, %1, off offset:16\n\t"
                 "global_load_async_to_lds_b128 %2, %3, off\n\t"
                 "global_load_async_to_lds_b128 %2, %3, off offset:16"
                 :: "v"(la), "v"(ga), "v"(lb), "v"(gb)
                 : "memory");
  };

  issue_tile(0, 0);
  issue_tile(1, 1);

  const int fr = lane & 15;      // row-in-tile / col-in-tile index
  const int hf = lane >> 4;      // K-half select per ISA lane layout

  const int KT = HID / BK;       // 64
  for (int kt = 0; kt < KT; ++kt) {
    const int cur = kt % NBUF;
    // drain this wave's async copies for tile kt (in-order: oldest 4 complete)
    if (kt + 1 < KT) asm volatile("s_wait_asynccnt 0x4" ::: "memory");
    else             asm volatile("s_wait_asynccnt 0x0" ::: "memory");
    // all of this wave's fragment reads from the previous iteration are done
    asm volatile("s_wait_dscnt 0x0" ::: "memory");
    __syncthreads();
    // refill the buffer that was consumed in the previous iteration (not `cur`)
    if (kt + 2 < KT) issue_tile(kt + 2, (kt + 2) % NBUF);

    const __bf16* baseA = &sA[cur][0];
    const __bf16* baseB = &sB[cur][0];
    AFrag a[4], b[2];
#pragma unroll
    for (int i = 0; i < 4; ++i) {
      // A 16x32 bf16: lanes 0-15 hold K 0..7 & 16..23; lanes 16-31 hold K 8..15 & 24..31
      const __bf16* p = baseA + (wm * 64 + i * 16 + fr) * LDT + hf * 8;
      a[i].q[0] = *(const u32x4*)p;
      a[i].q[1] = *(const u32x4*)(p + 16);
    }
#pragma unroll
    for (int j = 0; j < 2; ++j) {
      // B 32x16 bf16: lane = N, lanes 0-15 hold K 0..7 & 16..23; lanes 16-31 hold K 8..15 & 24..31
      const __bf16* p = baseB + (wn * 32 + j * 16 + fr) * LDT + hf * 8;
      b[j].q[0] = *(const u32x4*)p;
      b[j].q[1] = *(const u32x4*)(p + 16);
    }
#pragma unroll
    for (int i = 0; i < 4; ++i)
#pragma unroll
      for (int j = 0; j < 2; ++j)
        acc[i][j] = __builtin_amdgcn_wmma_f32_16x16x32_bf16(
            false, a[i].v, false, b[j].v, (short)0, acc[i][j], false, false);
  }

  // ---- fused LoRA epilogue: group g is uniform per block (BN=128 tiles
  // never straddle the 2048 / 2560 segment boundaries)
  int g, colseg, bstride;
  const float* Bmat;
  if (n0 < QSZ)             { g = 0; Bmat = Bq; bstride = QSZ;  colseg = 0;          }
  else if (n0 < QSZ + KVSZ) { g = 1; Bmat = Bk; bstride = KVSZ; colseg = QSZ;        }
  else                      { g = 2; Bmat = Bv; bstride = KVSZ; colseg = QSZ + KVSZ; }

  const int lr = lane >> 4, lc = lane & 15;
#pragma unroll
  for (int i = 0; i < 4; ++i) {
#pragma unroll
    for (int v = 0; v < 8; ++v) {
      const int row  = m0 + wm * 64 + i * 16 + lr * 8 + v;   // C-matrix lane layout
      const int slot = slots[row];
      const f32x4* H = (const f32x4*)(hbuf + (size_t)row * 48 + g * LRANK);
      const f32x4 h0 = H[0], h1 = H[1], h2 = H[2], h3 = H[3];
#pragma unroll
      for (int j = 0; j < 2; ++j) {
        const int col = n0 + wn * 32 + j * 16 + lc;
        const f32x4* Bp = (const f32x4*)(Bmat + ((size_t)slot * bstride + (col - colseg)) * LRANK);
        const f32x4 b0 = Bp[0], b1 = Bp[1], b2 = Bp[2], b3 = Bp[3];
        float d = h0.x * b0.x + h0.y * b0.y + h0.z * b0.z + h0.w * b0.w
                + h1.x * b1.x + h1.y * b1.y + h1.z * b1.z + h1.w * b1.w
                + h2.x * b2.x + h2.y * b2.y + h2.z * b2.z + h2.w * b2.w
                + h3.x * b3.x + h3.y * b3.y + h3.z * b3.z + h3.w * b3.w;
        out[(size_t)row * OUTN + col] = acc[i][j][v] + d;
      }
    }
  }
}

// ================= launcher =================
extern "C" void kernel_launch(void* const* d_in, const int* in_sizes, int n_in,
                              void* d_out, int out_size, void* d_ws, size_t ws_size,
                              hipStream_t stream) {
  const float* x       = (const float*)d_in[0];
  const float* weight  = (const float*)d_in[1];
  const float* loraA   = (const float*)d_in[2];
  const float* Bq      = (const float*)d_in[3];
  const float* Bk      = (const float*)d_in[4];
  const float* Bv      = (const float*)d_in[5];
  const float* scaling = (const float*)d_in[6];
  const int*   slots   = (const int*)d_in[7];
  float*       out     = (float*)d_out;

  // workspace layout: x_bf16 (32MB) | w_bf16 (12MB) | h_scaled (1.5MB)
  char* ws = (char*)d_ws;
  __bf16* xbf  = (__bf16*)ws;
  __bf16* wbf  = (__bf16*)(ws + (size_t)T_TOK * HID * sizeof(__bf16));
  float*  hbuf = (float*)(ws + (size_t)T_TOK * HID * sizeof(__bf16)
                             + (size_t)OUTN * HID * sizeof(__bf16));

  const int nx4 = T_TOK * HID / 4;
  const int nw4 = OUTN * HID / 4;
  cvt_bf16_kernel<<<(nx4 + 255) / 256, 256, 0, stream>>>(x, xbf, nx4);
  cvt_bf16_kernel<<<(nw4 + 255) / 256, 256, 0, stream>>>(weight, wbf, nw4);
  lora_h_kernel<<<T_TOK, 256, 0, stream>>>(x, loraA, scaling, slots, hbuf);

  dim3 grid(OUTN / BN, T_TOK / BM);
  qkv_gemm_kernel<<<grid, 256, 0, stream>>>(xbf, wbf, hbuf, Bq, Bk, Bv, slots, out);
}
